// Encoder_34583076667526
// MI455X (gfx1250) — matrix operand order repaired and checked by reading, not compiled, not run on hardware
//
#include <hip/hip_runtime.h>

// ---------------------------------------------------------------------------
// GRU encoder for MI455X (gfx1250, wave32, WMMA).
// B=256, T=512, E=256, H=256, K=E+H=512.
//  * batch rows independent -> 16 persistent WGs, zero cross-WG sync
//  * bf16 WMMA 16x16x32, f32 accumulate; weights repacked once into
//    fragment-native bf16 layout in workspace (768 KB, L2-resident)
//  * split LDS panels (x double-buffered | h | r*h) -> only 2 barriers/step
//  * 512 threads (16 wave32) per WG, one 16-col N-tile per wave
//  * per-step output streamed with global_store_async_from_lds_b128
//    (ASYNCcnt), overlapping the recurrence
// ---------------------------------------------------------------------------

#define B_  256
#define T_  512
#define E_  256
#define H_  256
#define K_  512
#define PSTRIDE 264   // bf16 elems per panel row (256 + pad, odd 16B units)
#define HSTRIDE 260   // f32 elems per hbuf row (256 + pad, bank-skewed)

typedef __attribute__((ext_vector_type(16))) __bf16 v16bf;
typedef __attribute__((ext_vector_type(8)))  float  v8f;
typedef __attribute__((ext_vector_type(4)))  int    i4;

union FragU { v16bf v; i4 q[2]; };

__device__ __forceinline__ unsigned short f2bf(float f) {
  union { float f; unsigned u; } x; x.f = f;
  unsigned u = x.u;
  u += 0x7FFFu + ((u >> 16) & 1u);          // round-to-nearest-even
  return (unsigned short)(u >> 16);
}

__device__ __forceinline__ float sigmoid_f(float x) {
  return 1.0f / (1.0f + __expf(-x));
}

// ---------------------------------------------------------------------------
// Prologue: repack Wz|Wr|Wm (f32 [H, K]) into bf16 B-fragment-native tiles.
// Fragment = 16 cols x 32 K, ISA B layout: lanes 0-15 hold K 0..7,16..23 of
// column n = lane, lanes 16-31 hold K 8..15,24..31. Storage: frag-major, 32 lanes x 16
// bf16 (32B/lane, b128-loadable). fragIndex = ((g*16 + nt)*16 + kt).
// ---------------------------------------------------------------------------
__global__ void gru_pack_weights(const float* __restrict__ Wz,
                                 const float* __restrict__ Wr,
                                 const float* __restrict__ Wm,
                                 unsigned short* __restrict__ wpack) {
  const int frag = blockIdx.x;              // 0..767
  const int lane = threadIdx.x;             // 0..31
  const int g  = frag >> 8;
  const int nt = (frag >> 4) & 15;
  const int kt = frag & 15;
  const float* W = (g == 0) ? Wz : (g == 1) ? Wr : Wm;
  const int n     = nt * 16 + (lane & 15);
  const int kbase = kt * 32 + ((lane >> 4) << 3);
  unsigned short* dst = wpack + (size_t)frag * 512 + lane * 16;
#pragma unroll
  for (int e = 0; e < 16; ++e)
    dst[e] = f2bf(W[(size_t)n * K_ + kbase + (e < 8 ? e : e + 8)]);
}

// ---------------------------------------------------------------------------
// Persistent recurrent kernel: one WG (16 wave32) per 16 batch rows.
// Wave wv owns output columns [wv*16, wv*16+16).
// ---------------------------------------------------------------------------
__global__ __launch_bounds__(512, 1)
void gru_persistent(const float* __restrict__ emb,          // [B, T, E]
                    const unsigned short* __restrict__ wpack,
                    const float* __restrict__ bz,
                    const float* __restrict__ br,
                    const float* __restrict__ bm,
                    float* __restrict__ out) {              // [B, T, H]
  __shared__ unsigned short xpan[2][16][PSTRIDE]; // bf16 x_t (double-buffered)
  __shared__ unsigned short hpan[16][PSTRIDE];    // bf16 h_{t-1}
  __shared__ unsigned short rpan[16][PSTRIDE];    // bf16 r*h
  __shared__ float          hbuf[16][HSTRIDE];    // f32 hidden state

  const int tid  = threadIdx.x;
  const int lane = tid & 31;
  const int wv   = tid >> 5;        // wave 0..15 -> N-tile wv
  const int mlo  = lane & 15;
  const int hsel = lane >> 4;       // half-wave select
  const int rb   = blockIdx.x * 16; // batch row base
  const int n    = wv * 16 + mlo;   // this lane's output column

  // h0 = 0
  for (int i = tid; i < 16 * H_; i += 512) {
    const int r = i >> 8, c = i & 255;
    hbuf[r][c] = 0.0f;
    hpan[r][c] = 0;                 // bf16(0)
  }
  __syncthreads();

  const float bzv = bz[n], brv = br[n], bmv = bm[n];

  // fragment streams (gate, ntile=wv): lane-contiguous 32B records, 1KB/ktile
  const unsigned short* pz = wpack + (size_t)((0 * 16 + wv) * 16) * 512 + lane * 16;
  const unsigned short* pr = wpack + (size_t)((1 * 16 + wv) * 16) * 512 + lane * 16;
  const unsigned short* pm = wpack + (size_t)((2 * 16 + wv) * 16) * 512 + lane * 16;

  float zsave[8];

  for (int t = 0; t < T_; ++t) {
    // ---- phase 1: load x_t -> bf16 xpan[t&1] ----
    {
      const int r  = tid >> 5;            // 0..15
      const int c0 = (tid & 31) * 8;      // 8 contiguous elems per thread
      const float* src = emb + ((size_t)(rb + r) * T_ + t) * E_ + c0;
      union { unsigned short s[8]; i4 q; } tmp;
#pragma unroll
      for (int j = 0; j < 8; ++j) tmp.s[j] = f2bf(src[j]);
      *(i4*)&xpan[t & 1][r][c0] = tmp.q;
      if (t + 1 < T_) __builtin_prefetch(src + E_, 0, 1);
    }
    __syncthreads();  // b1: x visible; h writes of step t-1 visible

    // ---- async-store h_{t-1} (this wave's own 16-column slice of hbuf) ----
    if (t > 0) {
#pragma unroll
      for (int s = 0; s < 2; ++s) {
        const int chunk = s * 32 + lane;       // 0..63
        const int r = chunk >> 2;              // row 0..15
        const int c = wv * 16 + (chunk & 3) * 4;
        unsigned ldsa = (unsigned)(uintptr_t)&hbuf[r][c];
        unsigned long long ga =
            (unsigned long long)(uintptr_t)(out + ((size_t)(rb + r) * T_ + (t - 1)) * H_ + c);
        asm volatile("global_store_async_from_lds_b128 %0, %1, off"
                     :: "v"(ga), "v"(ldsa) : "memory");
      }
    }

    // ---- phase 2: z/r GEMMs over K=512 ([x | h]) ----
    v8f az = {}, ar = {};
    const unsigned short* aX = &xpan[t & 1][mlo][0];
    const unsigned short* aH = &hpan[mlo][0];
#pragma unroll 4
    for (int kt = 0; kt < 8; ++kt) {          // K 0..255: x part
      const unsigned short* ap = aX + kt * 32 + hsel * 8;
      FragU a; a.q[0] = *(const i4*)ap; a.q[1] = *(const i4*)(ap + 16);
      FragU b0, b1;
      const unsigned short* q0 = pz + kt * 512;
      const unsigned short* q1 = pr + kt * 512;
      b0.q[0] = ((const i4*)q0)[0]; b0.q[1] = ((const i4*)q0)[1];
      b1.q[0] = ((const i4*)q1)[0]; b1.q[1] = ((const i4*)q1)[1];
      az = __builtin_amdgcn_wmma_f32_16x16x32_bf16(false, a.v, false, b0.v, (short)0, az, false, false);
      ar = __builtin_amdgcn_wmma_f32_16x16x32_bf16(false, a.v, false, b1.v, (short)0, ar, false, false);
    }
#pragma unroll 4
    for (int kt = 0; kt < 8; ++kt) {          // K 256..511: h part
      const unsigned short* ap = aH + kt * 32 + hsel * 8;
      FragU a; a.q[0] = *(const i4*)ap; a.q[1] = *(const i4*)(ap + 16);
      FragU b0, b1;
      const unsigned short* q0 = pz + (8 + kt) * 512;
      const unsigned short* q1 = pr + (8 + kt) * 512;
      b0.q[0] = ((const i4*)q0)[0]; b0.q[1] = ((const i4*)q0)[1];
      b1.q[0] = ((const i4*)q1)[0]; b1.q[1] = ((const i4*)q1)[1];
      az = __builtin_amdgcn_wmma_f32_16x16x32_bf16(false, a.v, false, b0.v, (short)0, az, false, false);
      ar = __builtin_amdgcn_wmma_f32_16x16x32_bf16(false, a.v, false, b1.v, (short)0, ar, false, false);
    }

    // ---- phase 3: z = sigm(.), write bf16(r*h) into rpan (private buffer,
    //      no reader until after b2 -> no barrier needed before writes) ----
#pragma unroll
    for (int j = 0; j < 8; ++j) {
      const int mr = j + hsel * 8;            // C/D layout: VGPR j -> row j(+8)
      zsave[j] = sigmoid_f(az[j] + bzv);
      const float rr = sigmoid_f(ar[j] + brv);
      rpan[mr][n] = f2bf(rr * hbuf[mr][n]);
    }
    __syncthreads();  // b2: r*h panel complete

    // ---- phase 4: candidate GEMM over K=512 ([x | r*h]) ----
    v8f am = {};
    const unsigned short* aR = &rpan[mlo][0];
#pragma unroll 4
    for (int kt = 0; kt < 8; ++kt) {          // x part
      const unsigned short* ap = aX + kt * 32 + hsel * 8;
      FragU a; a.q[0] = *(const i4*)ap; a.q[1] = *(const i4*)(ap + 16);
      FragU b0;
      const unsigned short* q0 = pm + kt * 512;
      b0.q[0] = ((const i4*)q0)[0]; b0.q[1] = ((const i4*)q0)[1];
      am = __builtin_amdgcn_wmma_f32_16x16x32_bf16(false, a.v, false, b0.v, (short)0, am, false, false);
    }
#pragma unroll 4
    for (int kt = 0; kt < 8; ++kt) {          // r*h part
      const unsigned short* ap = aR + kt * 32 + hsel * 8;
      FragU a; a.q[0] = *(const i4*)ap; a.q[1] = *(const i4*)(ap + 16);
      FragU b0;
      const unsigned short* q0 = pm + (8 + kt) * 512;
      b0.q[0] = ((const i4*)q0)[0]; b0.q[1] = ((const i4*)q0)[1];
      am = __builtin_amdgcn_wmma_f32_16x16x32_bf16(false, a.v, false, b0.v, (short)0, am, false, false);
    }

    // ---- phase 5: h_new = h + z*(tanh(cand)-h) ----
    // Wait for our own async stores of h_{t-1} (they read exactly the hbuf
    // elements this wave is about to overwrite; per-wave column ownership
    // means no cross-wave hazard).
    asm volatile("s_wait_asynccnt 0x0" ::: "memory");
#pragma unroll
    for (int j = 0; j < 8; ++j) {
      const int mr = j + hsel * 8;
      const float cand = tanhf(am[j] + bmv);
      const float h = hbuf[mr][n];
      const float hn = h + zsave[j] * (cand - h);   // (1-z)*h + z*cand
      hbuf[mr][n] = hn;                             // f32 state (+ output src)
      hpan[mr][n] = f2bf(hn);                       // next step's A panel
    }
    // next iteration's b1 makes h writes visible before the next z/r GEMM
  }

  // ---- final output row t = T-1 (reads only this wave's own columns) ----
#pragma unroll
  for (int s = 0; s < 2; ++s) {
    const int chunk = s * 32 + lane;
    const int r = chunk >> 2;
    const int c = wv * 16 + (chunk & 3) * 4;
    unsigned ldsa = (unsigned)(uintptr_t)&hbuf[r][c];
    unsigned long long ga =
        (unsigned long long)(uintptr_t)(out + ((size_t)(rb + r) * T_ + (T_ - 1)) * H_ + c);
    asm volatile("global_store_async_from_lds_b128 %0, %1, off"
                 :: "v"(ga), "v"(ldsa) : "memory");
  }
  // S_ENDPGM performs an implicit wait-idle, draining ASYNCcnt.
}

extern "C" void kernel_launch(void* const* d_in, const int* in_sizes, int n_in,
                              void* d_out, int out_size, void* d_ws, size_t ws_size,
                              hipStream_t stream) {
  const float* emb = (const float*)d_in[0];
  const float* Wz  = (const float*)d_in[1];
  const float* bz  = (const float*)d_in[2];
  const float* Wr  = (const float*)d_in[3];
  const float* br  = (const float*)d_in[4];
  const float* Wm  = (const float*)d_in[5];
  const float* bm  = (const float*)d_in[6];
  float* out = (float*)d_out;
  unsigned short* wpack = (unsigned short*)d_ws;  // 768 KB bf16 fragment store

  gru_pack_weights<<<768, 32, 0, stream>>>(Wz, Wr, Wm, wpack);
  gru_persistent<<<B_ / 16, 512, 0, stream>>>(emb, wpack, bz, br, bm, out);
}
